// sequence_to_pair_block_171798692293
// MI455X (gfx1250) — matrix operand order repaired and checked by reading, not compiled, not run on hardware
//
#include <hip/hip_runtime.h>
#include <cstddef>

// ---------------- problem constants ----------------
#define DIMD   1024
#define HEADS  16
#define PD     64
#define PP     512          // pooled sequence length P
#define BATCH  4
#define BPROWS (BATCH*PP)   // 2048 rows of x_ds
#define HC     (HEADS*PD)   // 1024

typedef __attribute__((ext_vector_type(16))) _Float16 v16h;
typedef __attribute__((ext_vector_type(8)))  float    v8f;
typedef __attribute__((ext_vector_type(4)))  float    v4f;

union FragAB { v16h v; unsigned u[8]; };

// ---- WMMA fragment loaders (CDNA5 ISA 7.12.2 layouts, wave32) ----
// A: 16x32 (MxK) f16. lane: m=l&15, hi=l>>4. VGPR v holds K pair at
//    k0 = (v>>2)*16 + hi*8 + (v&3)*2   (8 dwords -> two contiguous 16B -> b128)
__device__ __forceinline__ void load_frag_a(const _Float16* base, int ld, int lane, FragAB& f) {
    int m = lane & 15, hi = lane >> 4;
    const _Float16* rp = base + (size_t)m * ld;
#pragma unroll
    for (int v = 0; v < 8; ++v) {
        int k0 = ((v >> 2) << 4) + (hi << 3) + ((v & 3) << 1);
        f.u[v] = *reinterpret_cast<const unsigned*>(rp + k0);
    }
}
// B: 32x16 (KxN) f16 with B[k][n] = Bm[n][k] (Bm row-major N x K).
// lane: n=l&15, hi=l>>4. VGPR v holds K pair at k0 = (v>>2)*16 + hi*8 + (v&3)*2 (matches A).
__device__ __forceinline__ void load_frag_b(const _Float16* base, int ld, int lane, FragAB& f) {
    int n = lane & 15, hi = lane >> 4;
    const _Float16* rp = base + (size_t)n * ld;
#pragma unroll
    for (int v = 0; v < 8; ++v) {
        int k0 = ((v >> 2) << 4) + (hi << 3) + ((v & 3) << 1);
        f.u[v] = *reinterpret_cast<const unsigned*>(rp + k0);
    }
}

__device__ __forceinline__ v8f wmma_f16(const FragAB& a, const FragAB& b, v8f c) {
    return __builtin_amdgcn_wmma_f32_16x16x32_f16(false, a.v, false, b.v, (short)0, c, false, false);
}

// ---------------- kernel 1: avg-pool + gelu + f16 copy ----------------
// x is 32MB read exactly once -> NT loads, keep L2 for reused operands.
__global__ void k_down(const float* __restrict__ x, _Float16* __restrict__ xds16,
                       float* __restrict__ g) {
    int idx = blockIdx.x * 256 + threadIdx.x;          // BPROWS*DIMD threads
    int row = idx >> 10;                               // b*512 + i
    int col = idx & 1023;
    int b = row >> 9, i = row & 511;
    const float* xp = x + ((size_t)(b * 2048 + i * 4) * DIMD) + col;
    float x0 = __builtin_nontemporal_load(xp);
    float x1 = __builtin_nontemporal_load(xp + 1024);
    float x2 = __builtin_nontemporal_load(xp + 2048);
    float x3 = __builtin_nontemporal_load(xp + 3072);
    float s = 0.25f * (x0 + x1 + x2 + x3);
    xds16[idx] = (_Float16)s;
    g[idx] = 0.5f * s * (1.0f + erff(s * 0.70710678118654752f)); // exact gelu
}

// ---------------- kernel 2: f32 -> f16 convert ----------------
__global__ void k_cvt(const float* __restrict__ in, _Float16* __restrict__ out, int n) {
    int idx = blockIdx.x * 256 + threadIdx.x;
    if (idx < n) out[idx] = (_Float16)in[idx];
}

// ---------------- kernel 3: central-mask pos features (rows 0..511) ----------------
__global__ void k_posfeat(_Float16* __restrict__ pf) {
    int idx = blockIdx.x * 256 + threadIdx.x;          // 512*64
    int r = idx >> 6, f = idx & 63;
    float rel = (float)(r - (PP - 1));                 // rel_pos in [-511, 0]
    int i = f & 31;
    // widths[i] = i + 10**(i * log10(P-32+1)/31) = i + 481^(i/31)
    float width = (float)i + powf(481.0f, (float)i * (1.0f / 31.0f));
    float emb = (width > rel) ? 1.0f : 0.0f;
    float sgn = (rel > 0.f) ? 1.f : ((rel < 0.f) ? -1.f : 0.f);
    float v = (f < 32) ? emb : sgn * emb;
    pf[idx] = (_Float16)v;
}

// ---------------- kernel 4: WMMA GEMM  C = A(MxK) * Bm(NxK)^T ----------------
// out0 (optional): raw C as f16.  out1: scale*(C + bias[n]) as f16.
__global__ void k_gemm_f16(const _Float16* __restrict__ A, int lda,
                           const _Float16* __restrict__ Bm, int ldb, int K,
                           _Float16* __restrict__ out0, _Float16* __restrict__ out1,
                           const float* __restrict__ bias, float scale, int ldo) {
    int wave = threadIdx.x >> 5, lane = threadIdx.x & 31;
    int m0 = blockIdx.x * 16;
    int n0 = blockIdx.y * 64 + wave * 16;
    v8f acc = {0.f, 0.f, 0.f, 0.f, 0.f, 0.f, 0.f, 0.f};
    for (int k = 0; k < K; k += 32) {
        FragAB a, b;
        load_frag_a(A + (size_t)m0 * lda + k, lda, lane, a);
        load_frag_b(Bm + (size_t)n0 * ldb + k, ldb, lane, b);
        acc = wmma_f16(a, b, acc);
    }
    int n = lane & 15, hi = lane >> 4;
    int gn = n0 + n;
    float bv = bias[gn];
#pragma unroll
    for (int r = 0; r < 8; ++r) {
        int gm = m0 + r + hi * 8;                      // C layout: m = r + 8*hi
        float v = acc[r];
        if (out0) out0[(size_t)gm * ldo + gn] = (_Float16)v;
        out1[(size_t)gm * ldo + gn] = (_Float16)(scale * (v + bv));
    }
}

// ---------------- kernel 5: yq / yk projections (f32) ----------------
__global__ void k_yqyk(const float* __restrict__ g, const float* __restrict__ Wyq,
                       const float* __restrict__ Wyk, float* __restrict__ yq,
                       float* __restrict__ yk) {
    __shared__ float gs[DIMD];
    int row = blockIdx.x;                              // 0..2047
    const float* gp = g + (size_t)row * DIMD;
    for (int i = threadIdx.x; i < DIMD; i += 64) gs[i] = gp[i];
    __syncthreads();
    int d = threadIdx.x;                               // 0..63
    const float4* wq = (const float4*)(Wyq + (size_t)d * DIMD);
    const float4* wk = (const float4*)(Wyk + (size_t)d * DIMD);
    const float4* g4 = (const float4*)gs;
    float aq = 0.f, ak = 0.f;
    for (int k4 = 0; k4 < DIMD / 4; ++k4) {
        float4 gv = g4[k4], a = wq[k4], b = wk[k4];
        aq = fmaf(gv.x, a.x, aq); aq = fmaf(gv.y, a.y, aq);
        aq = fmaf(gv.z, a.z, aq); aq = fmaf(gv.w, a.w, aq);
        ak = fmaf(gv.x, b.x, ak); ak = fmaf(gv.y, b.y, ak);
        ak = fmaf(gv.z, b.z, ak); ak = fmaf(gv.w, b.w, ak);
    }
    yq[(size_t)row * PD + d] = aq;
    yk[(size_t)row * PD + d] = ak;
}

// ---------------- kernel 6: fused pair-block (the 256MB writer) ----------------
// One WG per (b, 16-i-tile, 16-j-tile). 4 waves x 4 heads. posf holds 0.5*pos_enc,
// so all three GEMMs chain into one WMMA accumulator. Output stored with NT hint:
// written once, never re-read; keeps the 17MB of reused f16 operands L2-resident.
__global__ void k_pair(const _Float16* __restrict__ qf, const _Float16* __restrict__ qbf,
                       const _Float16* __restrict__ kf, const _Float16* __restrict__ kbf,
                       const _Float16* __restrict__ posf,
                       const float* __restrict__ yq, const float* __restrict__ yk,
                       const float* __restrict__ Wout, const float* __restrict__ bout,
                       float* __restrict__ out) {
    __shared__ float a_lds[256 * 17];                  // [ij pos][head], padded
    __shared__ float woutT[HEADS * PD];                // [h][d]

    int b  = blockIdx.z;
    int i0 = blockIdx.y * 16, j0 = blockIdx.x * 16;
    int wave = threadIdx.x >> 5, lane = threadIdx.x & 31;

    for (int e = threadIdx.x; e < HEADS * PD; e += 128) {
        int d = e >> 4, h = e & 15;                    // Wout is [64][16] row-major
        woutT[h * PD + d] = Wout[e];
    }

    const _Float16* qrow  = qf  + (size_t)(b * PP + i0) * HC;
    const _Float16* qbrow = qbf + (size_t)(b * PP + i0) * HC;
    const _Float16* krow  = kf  + (size_t)(b * PP + j0) * HC;
    const _Float16* kbrow = kbf + (size_t)(b * PP + j0) * HC;
    const _Float16* pirow = posf + (size_t)i0 * HC;
    const _Float16* pjrow = posf + (size_t)j0 * HC;

    int n = lane & 15, hi = lane >> 4;
#pragma unroll
    for (int hh = 0; hh < 4; ++hh) {
        int h = wave * 4 + hh;
        int cb = h * PD;
        v8f acc = {0.f, 0.f, 0.f, 0.f, 0.f, 0.f, 0.f, 0.f};
#pragma unroll
        for (int kk = 0; kk < PD; kk += 32) {
            FragAB a, bfr;
            load_frag_a(qrow + cb + kk, HC, lane, a);      // q . k^T
            load_frag_b(krow + cb + kk, HC, lane, bfr);
            acc = wmma_f16(a, bfr, acc);
            load_frag_a(qbrow + cb + kk, HC, lane, a);     // (q+qb) . (0.5*pos)^T
            load_frag_b(pjrow + cb + kk, HC, lane, bfr);
            acc = wmma_f16(a, bfr, acc);
            load_frag_a(pirow + cb + kk, HC, lane, a);     // (0.5*pos) . (k+kb)^T
            load_frag_b(kbrow + cb + kk, HC, lane, bfr);
            acc = wmma_f16(a, bfr, acc);
        }
#pragma unroll
        for (int r = 0; r < 8; ++r) {
            int m = r + hi * 8;
            a_lds[(m * 16 + n) * 17 + h] = acc[r];
        }
    }
    __syncthreads();

    // head-mix + bout + yq + yk, write 64 contiguous f32 per (i,j) via NT b128 stores
    int t = threadIdx.x;
#pragma unroll
    for (int rep = 0; rep < 2; ++rep) {
        int p = t + rep * 128;
        int m = p >> 4, nn = p & 15;
        float ah[HEADS];
#pragma unroll
        for (int h = 0; h < HEADS; ++h) ah[h] = a_lds[p * 17 + h];
        const float4* yq4 = (const float4*)(yq + (size_t)(b * PP + i0 + m) * PD);
        const float4* yk4 = (const float4*)(yk + (size_t)(b * PP + j0 + nn) * PD);
        const float4* bo4 = (const float4*)bout;
        v4f* o4 = (v4f*)(out + ((size_t)(b * PP + i0 + m) * PP + (j0 + nn)) * PD);
        float res[PD];
#pragma unroll
        for (int d4 = 0; d4 < PD / 4; ++d4) {
            float4 s = bo4[d4], a = yq4[d4], bb = yk4[d4];
            res[4 * d4 + 0] = s.x + a.x + bb.x;
            res[4 * d4 + 1] = s.y + a.y + bb.y;
            res[4 * d4 + 2] = s.z + a.z + bb.z;
            res[4 * d4 + 3] = s.w + a.w + bb.w;
        }
#pragma unroll
        for (int h = 0; h < HEADS; ++h) {
            float av = ah[h];
            const float4* w4 = (const float4*)(&woutT[h * PD]);
#pragma unroll
            for (int d4 = 0; d4 < PD / 4; ++d4) {
                float4 w = w4[d4];
                res[4 * d4 + 0] = fmaf(av, w.x, res[4 * d4 + 0]);
                res[4 * d4 + 1] = fmaf(av, w.y, res[4 * d4 + 1]);
                res[4 * d4 + 2] = fmaf(av, w.z, res[4 * d4 + 2]);
                res[4 * d4 + 3] = fmaf(av, w.w, res[4 * d4 + 3]);
            }
        }
#pragma unroll
        for (int d4 = 0; d4 < PD / 4; ++d4) {
            v4f r4;
            r4.x = res[4 * d4 + 0]; r4.y = res[4 * d4 + 1];
            r4.z = res[4 * d4 + 2]; r4.w = res[4 * d4 + 3];
            __builtin_nontemporal_store(r4, o4 + d4);
        }
    }
}

// ---------------- host launcher ----------------
static inline char* bump(char*& p, size_t bytes) {
    char* r = p;
    p += (bytes + 255) & ~(size_t)255;
    return r;
}

extern "C" void kernel_launch(void* const* d_in, const int* in_sizes, int n_in,
                              void* d_out, int out_size, void* d_ws, size_t ws_size,
                              hipStream_t stream) {
    (void)in_sizes; (void)n_in; (void)out_size; (void)ws_size;
    const float* x     = (const float*)d_in[0];
    const float* Wq    = (const float*)d_in[1];
    const float* Wk    = (const float*)d_in[2];
    const float* Wpos  = (const float*)d_in[3];
    const float* bpos  = (const float*)d_in[4];
    const float* qbias = (const float*)d_in[5];   // [HC] per-n
    const float* kbias = (const float*)d_in[6];
    const float* Wout  = (const float*)d_in[7];
    const float* bout  = (const float*)d_in[8];
    const float* Wyq   = (const float*)d_in[9];
    const float* Wyk   = (const float*)d_in[10];
    float* out = (float*)d_out;

    char* p = (char*)d_ws;
    _Float16* xds16  = (_Float16*)bump(p, (size_t)BPROWS * DIMD * 2);
    _Float16* qf     = (_Float16*)bump(p, (size_t)BPROWS * HC * 2);
    _Float16* qbf    = (_Float16*)bump(p, (size_t)BPROWS * HC * 2);
    _Float16* kf     = (_Float16*)bump(p, (size_t)BPROWS * HC * 2);
    _Float16* kbf    = (_Float16*)bump(p, (size_t)BPROWS * HC * 2);
    _Float16* posf   = (_Float16*)bump(p, (size_t)PP * HC * 2);
    _Float16* Wq16   = (_Float16*)bump(p, (size_t)HC * DIMD * 2);
    _Float16* Wk16   = (_Float16*)bump(p, (size_t)HC * DIMD * 2);
    _Float16* Wpos16 = (_Float16*)bump(p, (size_t)HC * 64 * 2);
    _Float16* pf16   = (_Float16*)bump(p, (size_t)PP * 64 * 2);
    float*    g      = (float*)bump(p, (size_t)BPROWS * DIMD * 4);
    float*    yq     = (float*)bump(p, (size_t)BPROWS * PD * 4);
    float*    yk     = (float*)bump(p, (size_t)BPROWS * PD * 4);

    // 1) downsample + gelu + f16 cast (NT reads of x)
    k_down<<<(BPROWS * DIMD) / 256, 256, 0, stream>>>(x, xds16, g);
    // 2) weight conversions
    k_cvt<<<(HC * DIMD + 255) / 256, 256, 0, stream>>>(Wq, Wq16, HC * DIMD);
    k_cvt<<<(HC * DIMD + 255) / 256, 256, 0, stream>>>(Wk, Wk16, HC * DIMD);
    k_cvt<<<(HC * 64 + 255) / 256, 256, 0, stream>>>(Wpos, Wpos16, HC * 64);
    // 3) positional features (only rows 0..P-1 survive relative_shift)
    k_posfeat<<<(PP * 64) / 256, 256, 0, stream>>>(pf16);
    // 4) WMMA projections
    dim3 gq(BPROWS / 16, HC / 64);
    k_gemm_f16<<<gq, 128, 0, stream>>>(xds16, DIMD, Wq16, DIMD, DIMD, qf, qbf, qbias, 1.0f, HC);
    k_gemm_f16<<<gq, 128, 0, stream>>>(xds16, DIMD, Wk16, DIMD, DIMD, kf, kbf, kbias, 1.0f, HC);
    dim3 gp(PP / 16, HC / 64);   // posf = 0.5*(posfeat @ Wpos^T + bpos)
    k_gemm_f16<<<gp, 128, 0, stream>>>(pf16, 64, Wpos16, 64, 64, (_Float16*)nullptr, posf, bpos, 0.5f, HC);
    // 5) yq / yk
    k_yqyk<<<BPROWS, 64, 0, stream>>>(g, Wyq, Wyk, yq, yk);
    // 6) fused pair-block writer (NT output stores)
    dim3 gpair(PP / 16, PP / 16, BATCH);
    k_pair<<<gpair, 128, 0, stream>>>(qf, qbf, kf, kbf, posf, yq, yk, Wout, bout, out);
}
